// Discourse_62337155334343
// MI455X (gfx1250) — matrix operand order repaired and checked by reading, not compiled, not run on hardware
//
#include <hip/hip_runtime.h>
#include <stdint.h>

// Problem constants (match reference)
#define B_      64
#define E_      32
#define L_      512
#define D_      768
#define C_      1024
#define H_      8
#define EG_     32768
#define SPANH_  512
#define NLAB_   4
#define NNODE   (B_*E_)   // 2048

typedef __attribute__((ext_vector_type(16))) __bf16 v16bf;
typedef __attribute__((ext_vector_type(8)))  float  v8f;

union FragAB { v16bf v; unsigned int u[8]; };

__device__ __forceinline__ unsigned short f2bf(float f) {
  unsigned u = __float_as_uint(f);
  u += 0x7FFFu + ((u >> 16) & 1u);          // round-to-nearest-even
  return (unsigned short)(u >> 16);
}
__device__ __forceinline__ float bf2f(unsigned short h) {
  return __uint_as_float(((unsigned)h) << 16);
}
// order-preserving float<->uint encoding for atomicMax on floats
__device__ __forceinline__ unsigned f2ord(float f) {
  unsigned u = __float_as_uint(f);
  return (u & 0x80000000u) ? ~u : (u | 0x80000000u);
}
__device__ __forceinline__ float ord2f(unsigned u) {
  return __uint_as_float((u & 0x80000000u) ? (u & 0x7FFFFFFFu) : ~u);
}

// Async global->LDS DMA (CDNA5): contiguous 16B per lane, tracked by ASYNCcnt.
// Generic pointers to __shared__ carry the LDS offset in their low 32 bits.
__device__ __forceinline__ void async_ld_b128(unsigned lds_off,
                                              unsigned long long gaddr) {
  asm volatile("global_load_async_to_lds_b128 %0, %1, off"
               :: "v"(lds_off), "v"(gaddr) : "memory");
}

// ---------------------------------------------------------------- utilities
__global__ void k_cvt_bf16v(const float* __restrict__ in,
                            unsigned short* __restrict__ out, int n4) {
  for (int i = blockIdx.x * blockDim.x + threadIdx.x; i < n4;
       i += gridDim.x * blockDim.x) {
    const float4 f = ((const float4*)in)[i];
    const unsigned lo = ((unsigned)f2bf(f.y) << 16) | f2bf(f.x);
    const unsigned hi = ((unsigned)f2bf(f.w) << 16) | f2bf(f.z);
    ((uint2*)out)[i] = uint2{lo, hi};
  }
}

// convert f32 [R][Ccol] -> bf16 transposed [Ccol][R] (tiled, both sides coalesced)
__global__ __launch_bounds__(256)
void k_cvt_bf16_t(const float* __restrict__ in, unsigned short* __restrict__ out,
                  int R, int Ccol) {
  __shared__ unsigned short tile[32][33];
  const int r0 = blockIdx.y * 32, c0 = blockIdx.x * 32;
  for (int i = threadIdx.y; i < 32; i += 8)
    tile[i][threadIdx.x] = f2bf(in[(size_t)(r0 + i) * Ccol + c0 + threadIdx.x]);
  __syncthreads();
  for (int i = threadIdx.y; i < 32; i += 8)
    out[(size_t)(c0 + i) * R + r0 + threadIdx.x] = tile[threadIdx.x][i];
}

__global__ void k_fill_u32(unsigned int* __restrict__ p, unsigned int v, int n) {
  for (int i = blockIdx.x * blockDim.x + threadIdx.x; i < n;
       i += gridDim.x * blockDim.x)
    p[i] = v;
}

// ------------------------------------------------------- bf16 WMMA GEMM
// C(MxN,f32) = relu?( A(MxK,bf16 row-major) @ Bt(NxK,bf16 = B transposed) + bias )
// Requires M%64==0, N%128==0, K%32==0 (true for all call sites).
// Block: 256 threads = 8 waves; tile 64(M)x128(N); wave = 32x32 (2x2 WMMA tiles).
// Staging: async global->LDS b128 DMA, double buffered (3 async ops/wave/tile).
__global__ __launch_bounds__(256)
void k_gemm_bf16(const unsigned short* __restrict__ A,
                 const unsigned short* __restrict__ Bt,
                 const float* __restrict__ bias,
                 float* __restrict__ Cf,
                 int M, int N, int K, int relu)
{
  // Row stride 20 uints (16 data + 4 pad): keeps every b128 target 16B-aligned
  // and fragment reads bank-conflict-free (gcd(20,64)=4 -> 16 distinct banks).
  __shared__ unsigned int As[2][64 * 20];
  __shared__ unsigned int Bs[2][128 * 20];

  const int tid = threadIdx.x;
  const int nbx = N >> 7;
  const int m0  = (blockIdx.x / nbx) << 6;
  const int n0  = (blockIdx.x % nbx) << 7;

  const int lane = tid & 31, half = lane >> 4, mn = lane & 15;
  const int w = tid >> 5, wm = w >> 2, wn = w & 3;

  const unsigned int* Au = (const unsigned int*)A;   // bf16 pairs
  const unsigned int* Bu = (const unsigned int*)Bt;
  const int Kp = K >> 1;

  const v8f vzero = {0.f, 0.f, 0.f, 0.f, 0.f, 0.f, 0.f, 0.f};
  v8f acc[2][2];
  #pragma unroll
  for (int i = 0; i < 2; ++i)
    #pragma unroll
    for (int j = 0; j < 2; ++j) acc[i][j] = vzero;

  const int ar  = tid >> 2;            // 0..63 : A row / Bt row pass base
  const int ac4 = (tid & 3) << 2;      // uint column within tile row

  // issue one tile's staging (1x A b128 + 2x Bt b128 per thread)
  auto stage = [&](int buf, int kb) {
    async_ld_b128((unsigned)(size_t)&As[buf][ar * 20 + ac4],
                  (unsigned long long)(size_t)(Au + (size_t)(m0 + ar) * Kp +
                                               (kb >> 1) + ac4));
    #pragma unroll
    for (int p = 0; p < 2; ++p) {
      const int n = ar + p * 64;
      async_ld_b128((unsigned)(size_t)&Bs[buf][n * 20 + ac4],
                    (unsigned long long)(size_t)(Bu + (size_t)(n0 + n) * Kp +
                                                 (kb >> 1) + ac4));
    }
  };

  stage(0, 0);

  for (int kb = 0; kb < K; kb += 32) {
    const int buf = (kb >> 5) & 1;
    if (kb + 32 < K) {
      stage(buf ^ 1, kb + 32);                      // overlap next tile's DMA
      asm volatile("s_wait_asynccnt 0x3" ::: "memory");  // current tile done
    } else {
      asm volatile("s_wait_asynccnt 0x0" ::: "memory");
    }
    __syncthreads();

    // ---- fragments per documented VGPR layouts (wave32)
    FragAB af[2], bfm[2];
    #pragma unroll
    for (int i = 0; i < 2; ++i) {
      const int rbase = (wm * 32 + i * 16 + mn) * 20;
      #pragma unroll
      for (int v2 = 0; v2 < 8; ++v2) {
        // A 16x32 bf16: VGPR v<4 -> k-pair half*4+v ; v>=4 -> 8+half*4+(v-4)
        const int kk = (v2 < 4) ? (half * 4 + v2) : (8 + half * 4 + (v2 - 4));
        af[i].u[v2] = As[buf][rbase + kk];
      }
    }
    #pragma unroll
    for (int j = 0; j < 2; ++j) {
      const int cbase = (wn * 32 + j * 16 + mn) * 20;
      #pragma unroll
      for (int v2 = 0; v2 < 8; ++v2)  // B 32x16: same k-pair order as A (half*4+v ; 8+half*4+(v-4))
        bfm[j].u[v2] = Bs[buf][cbase + ((v2 < 4) ? (half * 4 + v2) : (8 + half * 4 + (v2 - 4)))];
    }

    #pragma unroll
    for (int i = 0; i < 2; ++i)
      #pragma unroll
      for (int j = 0; j < 2; ++j)
        acc[i][j] = __builtin_amdgcn_wmma_f32_16x16x32_bf16(
            false, af[i].v, false, bfm[j].v, (short)0, acc[i][j], false, false);
    __syncthreads();   // all reads of 'buf' done before it is restaged
  }

  // ---- epilogue: C/D layout — lane n=mn, VGPR p -> row half*8+p
  #pragma unroll
  for (int i = 0; i < 2; ++i)
    #pragma unroll
    for (int j = 0; j < 2; ++j)
      #pragma unroll
      for (int p = 0; p < 8; ++p) {
        const int r = m0 + wm * 32 + i * 16 + half * 8 + p;
        const int c = n0 + wn * 32 + j * 16 + mn;
        float vv = acc[i][j][p];
        if (bias) vv += bias[c];
        if (relu) vv = fmaxf(vv, 0.f);
        Cf[(size_t)r * N + c] = vv;
      }
}

// ------------------------------------------- att = Hid @ W2 + b2 (wave/row)
__global__ __launch_bounds__(256)
void k_att_reduce(const float* __restrict__ Hid, const float* __restrict__ W2,
                  const float* __restrict__ b2, float* __restrict__ att) {
  const int lane = threadIdx.x & 31;
  const int row  = blockIdx.x * 8 + (threadIdx.x >> 5);
  float acc = 0.f;
  for (int c = lane; c < SPANH_; c += 32)
    acc += Hid[(size_t)row * SPANH_ + c] * W2[c];
  #pragma unroll
  for (int off = 16; off; off >>= 1) acc += __shfl_xor(acc, off, 32);
  if (lane == 0) att[row] = acc + b2[0];
}

// ------------------ masked softmax over L + attention pooling, per (b,e)
__global__ __launch_bounds__(256)
void k_span_pool(const float* __restrict__ att, const float* __restrict__ mask,
                 const unsigned short* __restrict__ Xbf,
                 unsigned short* __restrict__ eduBF) {
  __shared__ float P[L_];
  __shared__ float red[256];
  const int be = blockIdx.x, b = be >> 5, tid = threadIdx.x;

  float l0 = att[b * L_ + tid]       - 1e5f * (1.f - mask[(size_t)be * L_ + tid]);
  float l1 = att[b * L_ + tid + 256] - 1e5f * (1.f - mask[(size_t)be * L_ + tid + 256]);
  red[tid] = fmaxf(l0, l1);
  __syncthreads();
  for (int s = 128; s; s >>= 1) {
    if (tid < s) red[tid] = fmaxf(red[tid], red[tid + s]);
    __syncthreads();
  }
  const float mmax = red[0];
  __syncthreads();
  const float e0 = __expf(l0 - mmax), e1 = __expf(l1 - mmax);
  P[tid] = e0; P[tid + 256] = e1;
  red[tid] = e0 + e1;
  __syncthreads();
  for (int s = 128; s; s >>= 1) {
    if (tid < s) red[tid] += red[tid + s];
    __syncthreads();
  }
  const float inv = 1.f / red[0];
  __syncthreads();

  #pragma unroll
  for (int dd = 0; dd < 3; ++dd) {           // D = 3*256
    const int d = tid + dd * 256;
    float acc = 0.f;
    for (int l = 0; l < L_; ++l)
      acc += P[l] * bf2f(Xbf[((size_t)b * L_ + l) * D_ + d]);
    eduBF[(size_t)be * D_ + d] = f2bf(acc * inv);
  }
}

// --------- collapse edge-attr GEMM:  u_t = Wr@eW, u_m = Wm@eW, u_b = (br+bm)@eW
__global__ void k_uvec(const float* __restrict__ eW, const float* __restrict__ wr,
                       const float* __restrict__ wm, const float* __restrict__ br,
                       const float* __restrict__ bm, float* __restrict__ u) {
  const int c = blockIdx.x * blockDim.x + threadIdx.x;
  float at = 0.f, am = 0.f, ab = 0.f;
  for (int j = 0; j < C_; ++j) {
    const float e = eW[(size_t)j * C_ + c];
    at += wr[j] * e;
    am += wm[j] * e;
    ab += (br[j] + bm[j]) * e;
  }
  u[c] = at; u[C_ + c] = am; u[2 * C_ + c] = ab;
}

// ---------------- edge attention pass 1: logits + segment max (wave per head)
__global__ __launch_bounds__(256)
void k_edge_logits(const int* __restrict__ edge, const float* __restrict__ etype,
                   const float* __restrict__ ismain, const float* __restrict__ Q,
                   const float* __restrict__ Kmat, const float* __restrict__ uv,
                   float* __restrict__ logits, unsigned int* __restrict__ maxb) {
  const int g = blockIdx.x;
  const int h = threadIdx.x >> 5, lane = threadIdx.x & 31;
  const int s = edge[g], d = edge[EG_ + g];
  const float t = etype[g], im = ismain[g];
  const int c = h * 128 + lane * 4;
  const float4 q4 = *(const float4*)(Q + (size_t)d * C_ + c);
  const float4 k4 = *(const float4*)(Kmat + (size_t)s * C_ + c);
  const float* ut = uv; const float* um = uv + C_; const float* ub = uv + 2 * C_;
  const float qa[4] = {q4.x, q4.y, q4.z, q4.w};
  const float ka[4] = {k4.x, k4.y, k4.z, k4.w};
  float p = 0.f;
  #pragma unroll
  for (int i = 0; i < 4; ++i)
    p += qa[i] * (ka[i] + t * ut[c + i] + im * um[c + i] + ub[c + i]);
  #pragma unroll
  for (int off = 16; off; off >>= 1) p += __shfl_xor(p, off, 32);
  if (lane == 0) {
    const float lg = p * 0.08838834764831845f;  // 1/sqrt(128)
    logits[g * H_ + h] = lg;
    atomicMax(&maxb[d * H_ + h], f2ord(lg));
  }
}

// ---------------- pass 2: exp + segment-sum denominator
__global__ void k_edge_exp(const int* __restrict__ edge,
                           const float* __restrict__ logits,
                           const unsigned int* __restrict__ maxb,
                           float* __restrict__ exb, float* __restrict__ den) {
  const int idx = blockIdx.x * blockDim.x + threadIdx.x;
  if (idx >= EG_ * H_) return;
  const int g = idx >> 3, h = idx & 7;
  const int d = edge[EG_ + g];
  const float ex = __expf(logits[idx] - ord2f(maxb[d * H_ + h]));
  exb[idx] = ex;
  atomicAdd(&den[d * H_ + h], ex);
}

// ---------------- pass 3: alpha * (v[src]+e)  scatter-add into agg (=skip)
__global__ __launch_bounds__(256)
void k_edge_agg(const int* __restrict__ edge, const float* __restrict__ etype,
                const float* __restrict__ ismain, const float* __restrict__ V,
                const float* __restrict__ uv, const float* __restrict__ exb,
                const float* __restrict__ den, float* __restrict__ agg) {
  const int g = blockIdx.x;
  const int s = edge[g], d = edge[EG_ + g];
  const float t = etype[g], im = ismain[g];
  const float* ut = uv; const float* um = uv + C_; const float* ub = uv + 2 * C_;
  #pragma unroll
  for (int r = 0; r < 4; ++r) {
    const int c = threadIdx.x + r * 256;
    const int h = c >> 7;
    const float alpha = exb[g * H_ + h] / den[d * H_ + h];
    const float val = V[(size_t)s * C_ + c] + t * ut[c] + im * um[c] + ub[c];
    atomicAdd(&agg[(size_t)d * C_ + c], alpha * val);
  }
}

// ---------------- final FC on last-EDU rows
__global__ void k_final(const float* __restrict__ node2, const int* __restrict__ lens,
                        const float* __restrict__ fcW, const float* __restrict__ fcb,
                        float* __restrict__ out) {
  const int t = threadIdx.x;          // 256 = 64 b * 4 labels
  const int b = t >> 2, lab = t & 3;
  const int row = b * E_ + (lens[b] - 1);
  float acc = fcb[lab];
  for (int c = 0; c < C_; ++c)
    acc += node2[(size_t)row * C_ + c] * fcW[c * NLAB_ + lab];
  out[b * NLAB_ + lab] = acc;
}

// =========================================================== host launcher
extern "C" void kernel_launch(void* const* d_in, const int* in_sizes, int n_in,
                              void* d_out, int out_size, void* d_ws, size_t ws_size,
                              hipStream_t stream) {
  (void)in_sizes; (void)n_in; (void)out_size; (void)ws_size;

  const float* last_hidden = (const float*)d_in[0];
  const float* mask   = (const float*)d_in[1];
  const int*   lens   = (const int*)d_in[2];
  const int*   edge   = (const int*)d_in[3];
  const float* etype  = (const float*)d_in[4];
  const float* ismain = (const float*)d_in[5];
  const float* W1 = (const float*)d_in[6];  const float* b1 = (const float*)d_in[7];
  const float* W2 = (const float*)d_in[8];  const float* b2 = (const float*)d_in[9];
  const float* Wr1 = (const float*)d_in[10]; const float* br1 = (const float*)d_in[11];
  const float* Wm1 = (const float*)d_in[12]; const float* bm1 = (const float*)d_in[13];
  const float* Wr2 = (const float*)d_in[14]; const float* br2 = (const float*)d_in[15];
  const float* Wm2 = (const float*)d_in[16]; const float* bm2 = (const float*)d_in[17];
  const float* q1W = (const float*)d_in[18]; const float* q1b = (const float*)d_in[19];
  const float* k1W = (const float*)d_in[20]; const float* k1b = (const float*)d_in[21];
  const float* v1W = (const float*)d_in[22]; const float* v1b = (const float*)d_in[23];
  const float* e1W = (const float*)d_in[24];
  const float* s1W = (const float*)d_in[25]; const float* s1b = (const float*)d_in[26];
  const float* q2W = (const float*)d_in[27]; const float* q2b = (const float*)d_in[28];
  const float* k2W = (const float*)d_in[29]; const float* k2b = (const float*)d_in[30];
  const float* v2W = (const float*)d_in[31]; const float* v2b = (const float*)d_in[32];
  const float* e2W = (const float*)d_in[33];
  const float* s2W = (const float*)d_in[34]; const float* s2b = (const float*)d_in[35];
  const float* fcW = (const float*)d_in[36]; const float* fcb = (const float*)d_in[37];
  float* out = (float*)d_out;

  uint8_t* ws = (uint8_t*)d_ws;
  size_t off = 0;
  auto carve = [&](size_t bytes) -> uint8_t* {
    uint8_t* p = ws + off;
    off += (bytes + 255) & ~(size_t)255;
    return p;
  };

  unsigned short* Xbf  = (unsigned short*)carve((size_t)B_ * L_ * D_ * 2);      // 48 MB
  uint8_t* big         = carve((size_t)B_ * L_ * SPANH_ * 4);                   // 64 MB (Hid, reused)
  unsigned short* W1t  = (unsigned short*)carve((size_t)D_ * SPANH_ * 2);
  float* att           = (float*)carve((size_t)B_ * L_ * 4);
  unsigned short* eduBF = (unsigned short*)carve((size_t)NNODE * D_ * 2);
  unsigned short* wq1 = (unsigned short*)carve((size_t)D_ * C_ * 2);   // transposed [C][D]
  unsigned short* wk1 = (unsigned short*)carve((size_t)D_ * C_ * 2);
  unsigned short* wv1 = (unsigned short*)carve((size_t)D_ * C_ * 2);
  unsigned short* ws1 = (unsigned short*)carve((size_t)D_ * C_ * 2);
  unsigned short* wq2 = (unsigned short*)carve((size_t)C_ * C_ * 2);   // transposed [C][C]
  unsigned short* wk2 = (unsigned short*)carve((size_t)C_ * C_ * 2);
  unsigned short* wv2 = (unsigned short*)carve((size_t)C_ * C_ * 2);
  unsigned short* ws2 = (unsigned short*)carve((size_t)C_ * C_ * 2);
  float* uv1    = (float*)carve(3 * C_ * 4);
  float* uv2    = (float*)carve(3 * C_ * 4);
  float* logits = (float*)carve((size_t)EG_ * H_ * 4);
  float* exb    = (float*)carve((size_t)EG_ * H_ * 4);
  unsigned int* maxb = (unsigned int*)carve((size_t)NNODE * H_ * 4);
  float* den    = (float*)carve((size_t)NNODE * H_ * 4);

  // Aliased views of the 64 MB region (Hid dead after k_att_reduce)
  float* Hid  = (float*)big;
  float* Qm   = (float*)big;                             // 8 MB
  float* Km   = (float*)(big + ((size_t)8  << 20));      // 8 MB
  float* Vm   = (float*)(big + ((size_t)16 << 20));      // 8 MB
  float* agg1 = (float*)(big + ((size_t)24 << 20));      // 8 MB
  unsigned short* node1bf = (unsigned short*)(big + ((size_t)32 << 20)); // 4 MB
  float* agg2 = (float*)(big + ((size_t)40 << 20));      // 8 MB

  const int TB = 256;
  const dim3 tb2(32, 8);

  // ---- conversions: activations straight, weights transposed to [N][K]
  k_cvt_bf16v<<<1024, TB, 0, stream>>>(last_hidden, Xbf, B_ * L_ * D_ / 4);
  k_cvt_bf16_t<<<dim3(SPANH_ / 32, D_ / 32), tb2, 0, stream>>>(W1,  W1t, D_, SPANH_);
  k_cvt_bf16_t<<<dim3(C_ / 32, D_ / 32), tb2, 0, stream>>>(q1W, wq1, D_, C_);
  k_cvt_bf16_t<<<dim3(C_ / 32, D_ / 32), tb2, 0, stream>>>(k1W, wk1, D_, C_);
  k_cvt_bf16_t<<<dim3(C_ / 32, D_ / 32), tb2, 0, stream>>>(v1W, wv1, D_, C_);
  k_cvt_bf16_t<<<dim3(C_ / 32, D_ / 32), tb2, 0, stream>>>(s1W, ws1, D_, C_);
  k_cvt_bf16_t<<<dim3(C_ / 32, C_ / 32), tb2, 0, stream>>>(q2W, wq2, C_, C_);
  k_cvt_bf16_t<<<dim3(C_ / 32, C_ / 32), tb2, 0, stream>>>(k2W, wk2, C_, C_);
  k_cvt_bf16_t<<<dim3(C_ / 32, C_ / 32), tb2, 0, stream>>>(v2W, wv2, C_, C_);
  k_cvt_bf16_t<<<dim3(C_ / 32, C_ / 32), tb2, 0, stream>>>(s2W, ws2, C_, C_);

  // ---- span MLP: Hid = relu(X@W1+b1); att = Hid@W2+b2; pooled EDU features
  k_gemm_bf16<<<(B_ * L_ / 64) * (SPANH_ / 128), TB, 0, stream>>>(
      Xbf, W1t, b1, Hid, B_ * L_, SPANH_, D_, 1);
  k_att_reduce<<<B_ * L_ / 8, TB, 0, stream>>>(Hid, W2, b2, att);
  k_span_pool<<<NNODE, TB, 0, stream>>>(att, mask, Xbf, eduBF);

  // ---- layer 1
  k_uvec<<<C_ / TB, TB, 0, stream>>>(e1W, Wr1, Wm1, br1, bm1, uv1);
  const int gproj = (NNODE / 64) * (C_ / 128);
  k_gemm_bf16<<<gproj, TB, 0, stream>>>(eduBF, wq1, q1b, Qm,   NNODE, C_, D_, 0);
  k_gemm_bf16<<<gproj, TB, 0, stream>>>(eduBF, wk1, k1b, Km,   NNODE, C_, D_, 0);
  k_gemm_bf16<<<gproj, TB, 0, stream>>>(eduBF, wv1, v1b, Vm,   NNODE, C_, D_, 0);
  k_gemm_bf16<<<gproj, TB, 0, stream>>>(eduBF, ws1, s1b, agg1, NNODE, C_, D_, 0);
  k_fill_u32<<<64, TB, 0, stream>>>(maxb, 0u, NNODE * H_);
  k_fill_u32<<<64, TB, 0, stream>>>((unsigned int*)den, 0u, NNODE * H_);
  k_edge_logits<<<EG_, TB, 0, stream>>>(edge, etype, ismain, Qm, Km, uv1, logits, maxb);
  k_edge_exp<<<EG_ * H_ / TB, TB, 0, stream>>>(edge, logits, maxb, exb, den);
  k_edge_agg<<<EG_, TB, 0, stream>>>(edge, etype, ismain, Vm, uv1, exb, den, agg1);
  k_cvt_bf16v<<<1024, TB, 0, stream>>>(agg1, node1bf, NNODE * C_ / 4);

  // ---- layer 2
  k_uvec<<<C_ / TB, TB, 0, stream>>>(e2W, Wr2, Wm2, br2, bm2, uv2);
  k_gemm_bf16<<<gproj, TB, 0, stream>>>(node1bf, wq2, q2b, Qm,   NNODE, C_, C_, 0);
  k_gemm_bf16<<<gproj, TB, 0, stream>>>(node1bf, wk2, k2b, Km,   NNODE, C_, C_, 0);
  k_gemm_bf16<<<gproj, TB, 0, stream>>>(node1bf, wv2, v2b, Vm,   NNODE, C_, C_, 0);
  k_gemm_bf16<<<gproj, TB, 0, stream>>>(node1bf, ws2, s2b, agg2, NNODE, C_, C_, 0);
  k_fill_u32<<<64, TB, 0, stream>>>(maxb, 0u, NNODE * H_);
  k_fill_u32<<<64, TB, 0, stream>>>((unsigned int*)den, 0u, NNODE * H_);
  k_edge_logits<<<EG_, TB, 0, stream>>>(edge, etype, ismain, Qm, Km, uv2, logits, maxb);
  k_edge_exp<<<EG_ * H_ / TB, TB, 0, stream>>>(edge, logits, maxb, exb, den);
  k_edge_agg<<<EG_, TB, 0, stream>>>(edge, etype, ismain, Vm, uv2, exb, den, agg2);

  // ---- final FC
  k_final<<<1, 256, 0, stream>>>(agg2, lens, fcW, fcb, out);
}
